// EarthSpecificBlock_4844723109908
// MI455X (gfx1250) — matrix operand order repaired and checked by reading, not compiled, not run on hardware
//
#include <hip/hip_runtime.h>
#include <math.h>

typedef __attribute__((ext_vector_type(16))) _Float16 v16h;
typedef __attribute__((ext_vector_type(8)))  _Float16 v8h;
typedef __attribute__((ext_vector_type(8)))  float    v8f;

#define DIM     192
#define HEADS   6
#define HD      32
#define N_TOK   144
#define NWIN    32           // NW
#define BNL     16           // B * NLON
#define NWIN_TOK (NWIN * N_TOK)     // 4608
#define M_TOTAL  (BNL * NWIN_TOK)   // 73728 tokens
#define L_TOT    (8 * 48 * 96)      // 36864
#define PART_SZ  ((size_t)M_TOTAL * DIM)   // q/k/v part size in halves (= 14,155,776)

union V16H { v16h v; v8h h[2]; };

// ---------- WMMA fragment loaders (CDNA5 wave32 layouts, CDNA5 ISA) ----------

// A-matrix 16x32 f16, row-major source with leading dim `ld` (halves).
// lane: M = lane&15 ; halves e<8 -> K = kbase + (lane>>4)*8 + e ; e>=8 -> K = kbase + 16 + (lane>>4)*8 + (e-8)
__device__ __forceinline__ v16h load_a_frag(const _Float16* base, int ld, int lane, int kbase) {
  int m = lane & 15, hh = lane >> 4;
  const _Float16* p = base + (size_t)m * ld + kbase + hh * 8;
  V16H r;
  r.h[0] = *(const v8h*)(p);
  r.h[1] = *(const v8h*)(p + 16);
  return r.v;
}

// B-matrix 32x16 f16 where B[k][n] = Wt[n][k] (Wt row-major, leading dim ld halves).
// lane: N = n0 + (lane&15) ; halves e<8 -> K = kbase + (lane>>4)*8 + e ; e>=8 -> K = kbase + 16 + (lane>>4)*8 + (e-8)
__device__ __forceinline__ v16h load_b_fragT(const _Float16* Wt, int ld, int n0, int kbase, int lane) {
  int n = n0 + (lane & 15);
  const _Float16* p = Wt + (size_t)n * ld + kbase + (lane >> 4) * 8;
  V16H r;
  r.h[0] = *(const v8h*)(p);
  r.h[1] = *(const v8h*)(p + 16);
  return r.v;
}

__device__ __forceinline__ v8f wmma32(v16h a, v16h b, v8f c) {
  return __builtin_amdgcn_wmma_f32_16x16x32_f16(false, a, false, b, (short)0, c, false, false);
}

// ---------- kernel 1: f32 -> f16 weight conversion ----------
__global__ __launch_bounds__(256) void convert_weights(
    const float* qkv_w, const float* proj_w, const float* fc1_w, const float* fc2_w,
    _Float16* qkv16, _Float16* proj16, _Float16* fc116, _Float16* fc216) {
  int i = blockIdx.x * 256 + threadIdx.x;
  if (i < 576 * 192) qkv16[i] = (_Float16)qkv_w[i];
  if (i < 192 * 192) proj16[i] = (_Float16)proj_w[i];
  if (i < 768 * 192) { fc116[i] = (_Float16)fc1_w[i]; fc216[i] = (_Float16)fc2_w[i]; }
}

// ---------- token (bw,w,n) -> source/destination flat index in x ----------
__device__ __forceinline__ size_t token_to_bl(int t) {
  int bw = t / NWIN_TOK;  int rem = t % NWIN_TOK;
  int w  = rem / N_TOK;   int n   = rem % N_TOK;
  int b   = bw >> 3,  lonb = bw & 7;
  int plb = w >> 3,   latb = w & 7;
  int zp = n / 72, r2 = n % 72, zl = r2 / 12, zo = r2 % 12;
  int pl = plb * 2 + zp, lat = latb * 6 + zl, lon = lonb * 12 + zo;
  int pls = (pl + 1) & 7;                 // roll by shift (1,3,6)
  int lats = lat + 3; if (lats >= 48) lats -= 48;
  int lons = lon + 6; if (lons >= 96) lons -= 96;
  return (size_t)b * L_TOT + (size_t)(pls * 48 + lats) * 96 + lons;
}

// ---------- kernel 2/6: LayerNorm (optionally fused with roll+partition), f32 -> f16 ----------
__global__ __launch_bounds__(128) void ln_kernel(const float* __restrict__ xin,
                                                 const float* __restrict__ gg,
                                                 const float* __restrict__ bb,
                                                 _Float16* __restrict__ out,
                                                 int partitioned) {
  int t = blockIdx.x * 4 + (threadIdx.x >> 5);
  int lane = threadIdx.x & 31;
  const float* src = partitioned ? (xin + token_to_bl(t) * DIM)
                                 : (xin + (size_t)t * DIM);
  float v[6];
#pragma unroll
  for (int j = 0; j < 6; ++j) v[j] = src[lane + 32 * j];
  float s = 0.f;
#pragma unroll
  for (int j = 0; j < 6; ++j) s += v[j];
#pragma unroll
  for (int m = 1; m < 32; m <<= 1) s += __shfl_xor(s, m, 32);
  float mean = s * (1.0f / 192.0f);
  float q = 0.f;
#pragma unroll
  for (int j = 0; j < 6; ++j) { float d = v[j] - mean; q += d * d; }
#pragma unroll
  for (int m = 1; m < 32; m <<= 1) q += __shfl_xor(q, m, 32);
  float rstd = rsqrtf(q * (1.0f / 192.0f) + 1e-5f);
  _Float16* dst = out + (size_t)t * DIM;
#pragma unroll
  for (int j = 0; j < 6; ++j) {
    int c = lane + 32 * j;
    dst[c] = (_Float16)((v[j] - mean) * rstd * gg[c] + bb[c]);
  }
}

// ---------- kernel 3: QKV GEMM [73728,192] x [576,192]^T, scatter to q/k/v ----------
// One block per 16-row tile; each wave owns 9 column tiles, processed 3-at-a-time
// with interleaved accumulators (covers the 5-slot F16 WMMA RAW hazard without NOPs).
__global__ __launch_bounds__(128) void qkv_gemm(const _Float16* __restrict__ H16,
                                                const _Float16* __restrict__ W16,
                                                const float* __restrict__ bias,
                                                _Float16* __restrict__ qkv) {
  __shared__ _Float16 As[16 * DIM];
  int tile = blockIdx.x;                    // 0..4607 (16-row tile, never crosses a window)
  int wv = threadIdx.x >> 5, lane = threadIdx.x & 31;
  const uint4* gsrc = (const uint4*)(H16 + (size_t)tile * 16 * DIM);
  uint4* ldst = (uint4*)As;
  for (int i = threadIdx.x; i < 16 * DIM / 8; i += 128) ldst[i] = gsrc[i];
  __syncthreads();
  v16h afr[6];
#pragma unroll
  for (int ks = 0; ks < 6; ++ks) afr[ks] = load_a_frag(As, DIM, lane, ks * 32);

  int token0 = tile * 16;
  int bw = token0 / NWIN_TOK;
  int w  = (token0 / N_TOK) % NWIN;
  int nb = token0 % N_TOK;
  int mbase = (lane >> 4) * 8, nc = lane & 15;

#pragma unroll
  for (int batch = 0; batch < 3; ++batch) {
    int ct0 = wv + 4 * (3 * batch);
    v8f acc[3];
#pragma unroll
    for (int j = 0; j < 3; ++j) acc[j] = (v8f){};
#pragma unroll
    for (int ks = 0; ks < 6; ++ks) {
#pragma unroll
      for (int j = 0; j < 3; ++j) {
        v16h b = load_b_fragT(W16, DIM, (ct0 + 4 * j) * 16, ks * 32, lane);
        acc[j] = wmma32(afr[ks], b, acc[j]);
      }
    }
#pragma unroll
    for (int j = 0; j < 3; ++j) {
      int colg = (ct0 + 4 * j) * 16 + nc;
      int part = colg / DIM;
      int cr = colg % DIM;
      int head = cr >> 5, d = cr & 31;
      float bv = bias[colg];
      float scale = (part == 0) ? 0.17677669529663687f : 1.0f;  // hd^-0.5 on q
      _Float16* dst = qkv + (size_t)part * PART_SZ
                    + ((((size_t)bw * HEADS + head) * NWIN + w) * N_TOK) * HD + d;
#pragma unroll
      for (int i = 0; i < 8; ++i) {
        int n = nb + mbase + i;
        dst[(size_t)n * HD] = (_Float16)((acc[j][i] + bv) * scale);
      }
    }
  }
}

// ---------- kernel 4: windowed attention (one 9-wave block per (bw,head,w)) ----------
__global__ __launch_bounds__(288) void attention_kernel(const _Float16* __restrict__ qkv,
                                                        const float* __restrict__ bias_table,
                                                        const float* __restrict__ attn_mask,
                                                        const int* __restrict__ pos_index,
                                                        _Float16* __restrict__ att_out) {
  __shared__ _Float16 qs [N_TOK * HD];     //  9216 B
  __shared__ _Float16 ksd[N_TOK * HD];     //  9216 B
  __shared__ _Float16 vT [HD * 160];       // 10240 B (V transposed, K padded 144->160)
  __shared__ _Float16 probs[N_TOK * 160];  // 46080 B (softmax probs, K padded)
  int blk  = blockIdx.x;
  int w    = blk & 31;
  int head = (blk >> 5) % HEADS;
  int bw   = blk / (NWIN * HEADS);
  int nlon = bw & 7;
  const _Float16* qg = qkv + ((((size_t)bw * HEADS + head) * NWIN + w) * N_TOK) * HD;
  const _Float16* kg = qg + PART_SZ;
  const _Float16* vg = qg + 2 * PART_SZ;
  int tid = threadIdx.x;
  // stage q,k (9216 B each, contiguous) and transposed/padded v
  for (int i = tid; i < 576; i += 288) {
    ((uint4*)qs)[i]  = ((const uint4*)qg)[i];
    ((uint4*)ksd)[i] = ((const uint4*)kg)[i];
  }
  for (int i = tid; i < N_TOK * HD; i += 288) {
    int tk = i >> 5, d = i & 31;
    vT[d * 160 + tk] = vg[i];
  }
  for (int i = tid; i < HD * 16; i += 288) {
    int d = i >> 4, tk = 144 + (i & 15);
    vT[d * 160 + tk] = (_Float16)0.0f;
  }
  __syncthreads();
  int wid = tid >> 5, lane = tid & 31;
  int mb = (lane >> 4) * 8, nc = lane & 15;
  // ---- scores: 16-row stripe x 144, K = 32 exactly -> 9 independent WMMAs ----
  v8f acc[9];
  v16h aq = load_a_frag(qs + wid * 16 * HD, HD, lane, 0);
#pragma unroll
  for (int j = 0; j < 9; ++j) {
    v16h bk = load_b_fragT(ksd, HD, j * 16, 0, lane);
    v8f z = {};
    acc[j] = wmma32(aq, bk, z);
  }
  // ---- bias + mask + row softmax (rows split across lane halves per C/D layout) ----
#pragma unroll
  for (int i = 0; i < 8; ++i) {
    int row = wid * 16 + mb + i;
    float vals[9];
    float mx = -1e30f;
#pragma unroll
    for (int j = 0; j < 9; ++j) {
      int col = j * 16 + nc;
      int idx = pos_index[row * N_TOK + col];
      float bsv = bias_table[((size_t)idx * NWIN + w) * HEADS + head];
      float msk = attn_mask[(((size_t)nlon * NWIN + w) * N_TOK + row) * N_TOK + col];
      float vt = acc[j][i] + bsv + msk;
      vals[j] = vt;
      mx = fmaxf(mx, vt);
    }
#pragma unroll
    for (int m2 = 1; m2 <= 8; m2 <<= 1) mx = fmaxf(mx, __shfl_xor(mx, m2, 32));
    float s = 0.f;
#pragma unroll
    for (int j = 0; j < 9; ++j) { vals[j] = __expf(vals[j] - mx); s += vals[j]; }
#pragma unroll
    for (int m2 = 1; m2 <= 8; m2 <<= 1) s += __shfl_xor(s, m2, 32);
    float inv = 1.0f / s;
#pragma unroll
    for (int j = 0; j < 9; ++j) probs[row * 160 + j * 16 + nc] = (_Float16)(vals[j] * inv);
    probs[row * 160 + 144 + nc] = (_Float16)0.0f;   // zero the K-pad columns
  }
  __syncthreads();
  // ---- PV: 16x160 @ 160x32 -> 16x32 ; shared A-fragment, 2 interleaved accumulators ----
  v8f o0 = (v8f){}, o1 = (v8f){};
#pragma unroll
  for (int ks = 0; ks < 5; ++ks) {
    v16h a  = load_a_frag(probs + wid * 16 * 160, 160, lane, ks * 32);
    v16h b0 = load_b_fragT(vT, 160, 0,  ks * 32, lane);
    v16h b1 = load_b_fragT(vT, 160, 16, ks * 32, lane);
    o0 = wmma32(a, b0, o0);
    o1 = wmma32(a, b1, o1);
  }
  size_t obase = (((size_t)bw * NWIN + w) * N_TOK + wid * 16 + mb) * DIM + head * HD + nc;
#pragma unroll
  for (int i = 0; i < 8; ++i) {
    att_out[obase + (size_t)i * DIM]      = (_Float16)o0[i];
    att_out[obase + (size_t)i * DIM + 16] = (_Float16)o1[i];
  }
}

// ---------- kernel 5: proj GEMM + reverse partition + roll back + residual ----------
// One block per 16-row tile; each wave owns 3 interleaved column tiles (12 total).
__global__ __launch_bounds__(128) void proj_residual(const _Float16* __restrict__ att,
                                                     const _Float16* __restrict__ W16,
                                                     const float* __restrict__ pbias,
                                                     const float* __restrict__ xin,
                                                     float* __restrict__ xnew) {
  __shared__ _Float16 As[16 * DIM];
  int tile = blockIdx.x;
  int wv = threadIdx.x >> 5, lane = threadIdx.x & 31;
  const uint4* gsrc = (const uint4*)(att + (size_t)tile * 16 * DIM);
  uint4* ldst = (uint4*)As;
  for (int i = threadIdx.x; i < 16 * DIM / 8; i += 128) ldst[i] = gsrc[i];
  __syncthreads();
  v16h afr[6];
#pragma unroll
  for (int ks = 0; ks < 6; ++ks) afr[ks] = load_a_frag(As, DIM, lane, ks * 32);
  int mbase = (lane >> 4) * 8, nc = lane & 15;

  v8f acc[3];
#pragma unroll
  for (int j = 0; j < 3; ++j) acc[j] = (v8f){};
#pragma unroll
  for (int ks = 0; ks < 6; ++ks) {
#pragma unroll
    for (int j = 0; j < 3; ++j) {
      v16h b = load_b_fragT(W16, DIM, (wv + 4 * j) * 16, ks * 32, lane);
      acc[j] = wmma32(afr[ks], b, acc[j]);
    }
  }
  // hoist the (roll-back) index decode: one per output row
  size_t rowbase[8];
#pragma unroll
  for (int i = 0; i < 8; ++i)
    rowbase[i] = token_to_bl(tile * 16 + mbase + i) * DIM;
#pragma unroll
  for (int j = 0; j < 3; ++j) {
    int colg = (wv + 4 * j) * 16 + nc;
    float bv = pbias[colg];
#pragma unroll
    for (int i = 0; i < 8; ++i) {
      size_t idx = rowbase[i] + colg;
      xnew[idx] = xin[idx] + acc[j][i] + bv;
    }
  }
}

// ---------- kernel 7: fused MLP fc1+GELU (LDS) -> fc2 + residual ----------
__global__ __launch_bounds__(128) void mlp_fused(const _Float16* __restrict__ H16,
                                                 const _Float16* __restrict__ W1,
                                                 const float* __restrict__ b1,
                                                 const _Float16* __restrict__ W2,
                                                 const float* __restrict__ b2,
                                                 const float* __restrict__ xnew,
                                                 float* __restrict__ out) {
  __shared__ _Float16 As[16 * DIM];        //  6144 B
  __shared__ _Float16 inter[16 * 768];     // 24576 B
  int tile = blockIdx.x;
  int wv = threadIdx.x >> 5, lane = threadIdx.x & 31;
  const uint4* gsrc = (const uint4*)(H16 + (size_t)tile * 16 * DIM);
  uint4* ldst = (uint4*)As;
  for (int i = threadIdx.x; i < 16 * DIM / 8; i += 128) ldst[i] = gsrc[i];
  __syncthreads();
  v16h afr[6];
#pragma unroll
  for (int ks = 0; ks < 6; ++ks) afr[ks] = load_a_frag(As, DIM, lane, ks * 32);
  int mbase = (lane >> 4) * 8, nc = lane & 15;

  // fc1 + exact GELU -> LDS ; 12 column tiles per wave, 3 interleaved at a time
#pragma unroll
  for (int batch = 0; batch < 4; ++batch) {
    int ct0 = wv + 4 * (3 * batch);
    v8f acc[3];
#pragma unroll
    for (int j = 0; j < 3; ++j) acc[j] = (v8f){};
#pragma unroll
    for (int ks = 0; ks < 6; ++ks) {
#pragma unroll
      for (int j = 0; j < 3; ++j) {
        v16h b = load_b_fragT(W1, DIM, (ct0 + 4 * j) * 16, ks * 32, lane);
        acc[j] = wmma32(afr[ks], b, acc[j]);
      }
    }
#pragma unroll
    for (int j = 0; j < 3; ++j) {
      int colg = (ct0 + 4 * j) * 16 + nc;
      float bv = b1[colg];
#pragma unroll
      for (int i = 0; i < 8; ++i) {
        float v = acc[j][i] + bv;
        v = 0.5f * v * (1.0f + erff(v * 0.70710678118654752f));
        inter[(mbase + i) * 768 + colg] = (_Float16)v;
      }
    }
  }
  __syncthreads();

  // fc2 + residual -> d_out ; 3 interleaved column tiles, shared A-fragment per K step
  v8f acc[3];
#pragma unroll
  for (int j = 0; j < 3; ++j) acc[j] = (v8f){};
#pragma unroll
  for (int ks = 0; ks < 24; ++ks) {
    v16h a = load_a_frag(inter, 768, lane, ks * 32);
#pragma unroll
    for (int j = 0; j < 3; ++j) {
      v16h b = load_b_fragT(W2, 768, (wv + 4 * j) * 16, ks * 32, lane);
      acc[j] = wmma32(a, b, acc[j]);
    }
  }
#pragma unroll
  for (int j = 0; j < 3; ++j) {
    int colg = (wv + 4 * j) * 16 + nc;
    float bv = b2[colg];
#pragma unroll
    for (int i = 0; i < 8; ++i) {
      size_t idx = (size_t)(tile * 16 + mbase + i) * DIM + colg;
      out[idx] = xnew[idx] + acc[j][i] + bv;
    }
  }
}

extern "C" void kernel_launch(void* const* d_in, const int* in_sizes, int n_in,
                              void* d_out, int out_size, void* d_ws, size_t ws_size,
                              hipStream_t stream) {
  (void)in_sizes; (void)n_in; (void)out_size; (void)ws_size;
  const float* x          = (const float*)d_in[0];
  const float* n1g        = (const float*)d_in[1];
  const float* n1b        = (const float*)d_in[2];
  const float* qkv_w      = (const float*)d_in[3];
  const float* qkv_b      = (const float*)d_in[4];
  const float* bias_table = (const float*)d_in[5];
  const float* proj_w     = (const float*)d_in[6];
  const float* proj_b     = (const float*)d_in[7];
  const float* n2g        = (const float*)d_in[8];
  const float* n2b        = (const float*)d_in[9];
  const float* fc1_w      = (const float*)d_in[10];
  const float* fc1_b      = (const float*)d_in[11];
  const float* fc2_w      = (const float*)d_in[12];
  const float* fc2_b      = (const float*)d_in[13];
  const float* attn_mask  = (const float*)d_in[14];
  const int*   pos_index  = (const int*)d_in[15];

  // workspace carve-up (halves); hot per-phase working sets are L2-resident (192 MB L2)
  _Float16* ws16 = (_Float16*)d_ws;
  size_t o = 0;
  _Float16* H16  = ws16 + o; o += PART_SZ;          // LN'd/partitioned activations (reused for LN2)
  _Float16* QKV  = ws16 + o; o += 3 * PART_SZ;      // q,k,v per-head layout
  _Float16* ATT  = ws16 + o; o += PART_SZ;          // attention output (window order)
  _Float16* QW16 = ws16 + o; o += 576 * 192;
  _Float16* PW16 = ws16 + o; o += 192 * 192;
  _Float16* F1W  = ws16 + o; o += 768 * 192;
  _Float16* F2W  = ws16 + o; o += 192 * 768;
  o = (o + 7) & ~(size_t)7;                         // 16B-align the f32 region
  float* XNEW = (float*)(ws16 + o);                 // residual x after attention (orig order)

  convert_weights<<<576, 256, 0, stream>>>(qkv_w, proj_w, fc1_w, fc2_w, QW16, PW16, F1W, F2W);
  ln_kernel<<<M_TOTAL / 4, 128, 0, stream>>>(x, n1g, n1b, H16, 1);
  qkv_gemm<<<M_TOTAL / 16, 128, 0, stream>>>(H16, QW16, qkv_b, QKV);
  attention_kernel<<<BNL * HEADS * NWIN, 288, 0, stream>>>(QKV, bias_table, attn_mask, pos_index, ATT);
  proj_residual<<<M_TOTAL / 16, 128, 0, stream>>>(ATT, PW16, proj_b, x, XNEW);
  ln_kernel<<<M_TOTAL / 4, 128, 0, stream>>>(XNEW, n2g, n2b, H16, 0);
  mlp_fused<<<M_TOTAL / 16, 128, 0, stream>>>(H16, F1W, fc1_b, F2W, fc2_b, XNEW, (float*)d_out);
}
